// SparseRouter_1915555414025
// MI455X (gfx1250) — matrix operand order repaired and checked by reading, not compiled, not run on hardware
//
#include <hip/hip_runtime.h>
#include <hip/hip_bf16.h>

// ---- problem constants (from reference) ----
#define BB 4
#define SS 8192
#define DD 2048
#define EE 64
#define KK 2
#define NTOK (BB * SS)          // 32768 tokens
#define TOK_PER_BLK 32          // 2 token tiles of 16
#define LDS_STRIDE (EE + 4)     // 68 floats: 16B-aligned rows, no 64-bank stride

typedef __attribute__((ext_vector_type(16))) __bf16 v16bf;
typedef __attribute__((ext_vector_type(8)))  float  v8f;

// ------------------------------------------------------------------
// Kernel 0: zero the reduction workspace (f hist @ ws[0..63], p sums @ ws[64..127])
// ------------------------------------------------------------------
__global__ void router_zero_ws(float* __restrict__ ws) {
    int i = threadIdx.x;
    if (i < 2 * EE) ws[i] = 0.0f;
}

// ------------------------------------------------------------------
// helper: pack 8 f32 (two float4) into bf16 lanes [base..base+7] of a v16bf
// ------------------------------------------------------------------
__device__ __forceinline__ void cvt8_to_bf16(v16bf& v, int base, float4 p, float4 q) {
    v[base + 0] = (__bf16)p.x;  v[base + 1] = (__bf16)p.y;
    v[base + 2] = (__bf16)p.z;  v[base + 3] = (__bf16)p.w;
    v[base + 4] = (__bf16)q.x;  v[base + 5] = (__bf16)q.y;
    v[base + 6] = (__bf16)q.z;  v[base + 7] = (__bf16)q.w;
}

// load one A (16x32) + one B (32x16) bf16 fragment for K-chunk at `kk`
__device__ __forceinline__ void load_frags(const float* __restrict__ xrow,
                                           const float* __restrict__ wrow,
                                           int kk, int hi, v16bf& a, v16bf& b) {
    // A layout: lane(m,hi): VGPR0..3 <- K = 8*hi+0..7 ; VGPR4..7 <- K = 16+8*hi+0..7
    const float4* a0 = (const float4*)(xrow + kk + 8 * hi);
    const float4* a1 = (const float4*)(xrow + kk + 16 + 8 * hi);
    float4 fa0 = a0[0], fa1 = a0[1];
    float4 fa2 = a1[0], fa3 = a1[1];
    // B layout: lane(n,hi): VGPR0..3 <- K = 8*hi+0..7 ; VGPR4..7 <- K = 16+8*hi+0..7
    const float4* b0 = (const float4*)(wrow + kk + 8 * hi);
    const float4* b1 = (const float4*)(wrow + kk + 16 + 8 * hi);
    float4 fb0 = b0[0], fb1 = b0[1], fb2 = b1[0], fb3 = b1[1];
    cvt8_to_bf16(a, 0, fa0, fa1);
    cvt8_to_bf16(a, 8, fa2, fa3);
    cvt8_to_bf16(b, 0, fb0, fb1);
    cvt8_to_bf16(b, 8, fb2, fb3);
}

// ------------------------------------------------------------------
// Kernel 1: logits GEMM via v_wmma_f32_16x16x32_bf16 + top-2/softmax/reductions
// ------------------------------------------------------------------
__global__ __launch_bounds__(256) void router_main(
        const float* __restrict__ x,        // [NTOK, DD] row-major
        const float* __restrict__ W,        // [EE, DD]  row-major
        float* __restrict__ out_w,          // [NTOK, 2]
        float* __restrict__ out_idx,        // [NTOK, 2] (indices stored as float)
        float* __restrict__ ws_f,           // [EE] top-1 counts
        float* __restrict__ ws_p)           // [EE] softmax-prob sums
{
    __shared__ float logits[TOK_PER_BLK][LDS_STRIDE];
    __shared__ float fhist[EE];

    const int tid        = threadIdx.x;
    const int wave       = tid >> 5;
    const int lane       = tid & 31;
    const int hi         = lane >> 4;       // half-wave select
    const int lane16     = lane & 15;
    const int expertTile = wave & 3;        // 4 tiles cover E=64
    const int tokenTile  = wave >> 2;       // 2 tiles cover 32 tokens

    const int gtoken = blockIdx.x * TOK_PER_BLK + tokenTile * 16 + lane16;
    const int expert = expertTile * 16 + lane16;

    const float* __restrict__ xrow = x + (size_t)gtoken * DD;
    const float* __restrict__ wrow = W + (size_t)expert * DD;

    // Two independent accumulator chains -> two WMMAs in flight on the XDL pipe.
    v8f acc0 = {}, acc1 = {};

    for (int kk = 0; kk < DD; kk += 64) {
        // stream-prefetch x ~2KB ahead (speculative; dropped if OOB)
        __builtin_prefetch(xrow + kk + 512, 0, 1);

        v16bf a0, b0, a1, b1;
        load_frags(xrow, wrow, kk,      hi, a0, b0);
        load_frags(xrow, wrow, kk + 32, hi, a1, b1);

        acc0 = __builtin_amdgcn_wmma_f32_16x16x32_bf16(
                false, a0, false, b0, (short)0, acc0, false, false);
        acc1 = __builtin_amdgcn_wmma_f32_16x16x32_bf16(
                false, a1, false, b1, (short)0, acc1, false, false);
    }
    v8f acc = acc0 + acc1;

    // C/D layout: VGPR r, lane L -> (m = r + 8*hi, n = lane16)
    #pragma unroll
    for (int r = 0; r < 8; ++r) {
        logits[tokenTile * 16 + r + 8 * hi][expertTile * 16 + lane16] = acc[r];
    }
    if (tid < EE) fhist[tid] = 0.0f;
    __syncthreads();

    // ---- per-token top-2, softmax weights, full-softmax probs (branchless) ----
    if (tid < TOK_PER_BLK) {
        const int t  = tid;
        const int gt = blockIdx.x * TOK_PER_BLK + t;

        float m1 = -3.402823466e+38f; int i1 = 0;
        float m2 = -3.402823466e+38f; int i2 = 0;
        // branchless top-2: pure selects -> v_cndmask, no EXEC games
        auto upd = [&](float v, int e) {
            bool  b1  = v > m1;
            bool  b2  = v > m2;
            float m2n = b1 ? m1 : (b2 ? v : m2);
            int   i2n = b1 ? i1 : (b2 ? e : i2);
            m1 = b1 ? v : m1;
            i1 = b1 ? e : i1;
            m2 = m2n; i2 = i2n;
        };
        #pragma unroll
        for (int e = 0; e < EE; e += 4) {
            float4 v = *(const float4*)&logits[t][e];   // ds_load_b128
            upd(v.x, e + 0); upd(v.y, e + 1);
            upd(v.z, e + 2); upd(v.w, e + 3);
        }

        // softmax over the top-2 logits
        float t2 = __expf(m2 - m1);
        float w1 = 1.0f / (1.0f + t2);
        out_w[(size_t)gt * 2 + 0] = w1;
        out_w[(size_t)gt * 2 + 1] = t2 * w1;
        out_idx[(size_t)gt * 2 + 0] = (float)i1;
        out_idx[(size_t)gt * 2 + 1] = (float)i2;

        // full softmax over all 64 experts (for p_i); store exp() in place
        float sum = 0.0f;
        #pragma unroll
        for (int e = 0; e < EE; e += 4) {
            float4 v = *(const float4*)&logits[t][e];
            v.x = __expf(v.x - m1); v.y = __expf(v.y - m1);
            v.z = __expf(v.z - m1); v.w = __expf(v.w - m1);
            sum += (v.x + v.y) + (v.z + v.w);
            *(float4*)&logits[t][e] = v;                // ds_store_b128
        }
        float inv = 1.0f / sum;
        #pragma unroll
        for (int e = 0; e < EE; e += 4) {
            float4 v = *(const float4*)&logits[t][e];
            v.x *= inv; v.y *= inv; v.z *= inv; v.w *= inv;
            *(float4*)&logits[t][e] = v;
        }

        atomicAdd(&fhist[i1], 1.0f);
    }
    __syncthreads();

    // ---- flush per-block partials: 64 threads, one expert column each ----
    if (tid < EE) {
        float s = 0.0f;
        #pragma unroll 8
        for (int t = 0; t < TOK_PER_BLK; ++t) s += logits[t][tid];
        atomicAdd(&ws_p[tid], s);
        float fh = fhist[tid];
        if (fh != 0.0f) atomicAdd(&ws_f[tid], fh);
    }
}

// ------------------------------------------------------------------
// Kernel 2: aux_loss = E * sum_e (f_e/n) * (p_e/n)
// ------------------------------------------------------------------
__global__ void router_finalize(const float* __restrict__ ws_f,
                                const float* __restrict__ ws_p,
                                float* __restrict__ out_aux)
{
    __shared__ float red[EE];
    int e = threadIdx.x;
    red[e] = ws_f[e] * ws_p[e];
    __syncthreads();
    if (e == 0) {
        float s = 0.0f;
        #pragma unroll
        for (int i = 0; i < EE; ++i) s += red[i];
        const float n = (float)NTOK;
        out_aux[0] = (float)EE * s / (n * n);
    }
}

// ------------------------------------------------------------------
extern "C" void kernel_launch(void* const* d_in, const int* in_sizes, int n_in,
                              void* d_out, int out_size, void* d_ws, size_t ws_size,
                              hipStream_t stream) {
    (void)in_sizes; (void)n_in; (void)out_size; (void)ws_size;

    const float* x = (const float*)d_in[0];   // [B,S,D] f32
    const float* W = (const float*)d_in[1];   // [E,D]   f32

    float* out     = (float*)d_out;
    float* out_w   = out;                         // 65536 weights
    float* out_idx = out + (size_t)NTOK * KK;     // 65536 indices
    float* out_aux = out + (size_t)2 * NTOK * KK; // 1 scalar

    float* ws   = (float*)d_ws;
    float* ws_f = ws;          // [64]
    float* ws_p = ws + EE;     // [64]

    router_zero_ws<<<1, 128, 0, stream>>>(ws);
    router_main<<<NTOK / TOK_PER_BLK, 256, 0, stream>>>(x, W, out_w, out_idx, ws_f, ws_p);
    router_finalize<<<1, EE, 0, stream>>>(ws_f, ws_p, out_aux);
}
